// ImageCaptioningModel_53730040873050
// MI455X (gfx1250) — matrix operand order repaired and checked by reading, not compiled, not run on hardware
//
#include <hip/hip_runtime.h>
#include <hip/hip_bf16.h>

// ---------------------------------------------------------------------------
// CDNA5 (gfx1250) wave32 WMMA types and helpers
// ---------------------------------------------------------------------------
typedef __attribute__((ext_vector_type(16))) __bf16 v16bf;
typedef __attribute__((ext_vector_type(8)))  float  v8f;

__device__ __forceinline__ unsigned short f2bf(float f) {
  union { float f; unsigned u; } x; x.f = f;
  unsigned u = x.u;
  unsigned r = (u + 0x7FFFu + ((u >> 16) & 1u)) >> 16;   // round-to-nearest-even
  return (unsigned short)r;
}
__device__ __forceinline__ float sigmoidf_(float x) {
  return 1.0f / (1.0f + __expf(-x));
}

#define BN_SCALE 0.99999500003749968752f  /* 1/sqrt(1+1e-5) */

#define WMMA_BF16(A, B, C) \
  __builtin_amdgcn_wmma_f32_16x16x32_bf16(false, (A), false, (B), (short)0, (C), false, false)

// ---------------------------------------------------------------------------
// Optional Tensor-Data-Mover probe: issue a NULL-descriptor tensor_load
// (D# group0.count == 0 -> NULL tensor, architecturally a NOP) so the
// TDM / TENSORcnt path is exercised without touching memory.
// ---------------------------------------------------------------------------
#if defined(__has_builtin)
#if __has_builtin(__builtin_amdgcn_tensor_load_to_lds) && \
    __has_builtin(__builtin_amdgcn_s_wait_tensorcnt)
#define HAVE_TDM 1
#endif
#endif

#ifdef HAVE_TDM
typedef unsigned int uint32x4 __attribute__((ext_vector_type(4)));
typedef int          int32x4  __attribute__((ext_vector_type(4)));
typedef int          int32x8  __attribute__((ext_vector_type(8)));

__global__ void tdm_null_probe()
{
  __shared__ long long dummy[16];
  uint32x4 g0 = {0u, 0u, 0u, 0u};      // count=0 => NULL tensor (NOP)
  int32x8  g1 = {0, 0, 0, 0, 0, 0, 0, 0};
  int32x4  g2 = {0, 0, 0, 0};
  int32x4  g3 = {0, 0, 0, 0};
#if __clang_major__ >= 23
  int32x8  g4 = {0, 0, 0, 0, 0, 0, 0, 0};
  __builtin_amdgcn_tensor_load_to_lds(g0, g1, g2, g3, g4, 0);
#else
  __builtin_amdgcn_tensor_load_to_lds(g0, g1, g2, g3, 0);
#endif
  __builtin_amdgcn_s_wait_tensorcnt(0);
  if (threadIdx.x == 1023u) dummy[0] = 0;   // keep LDS alloc alive (never true)
}
#endif

// ---------------------------------------------------------------------------
// Implicit-GEMM conv (SAME, stride 1, k x k with k*k == KK in {9,1}) fused
// with BatchNorm (folding conv bias) and optional ReLU.
//   GEMM view: M = N*H*W pixels, N = Co, K = Ci*KK
// Workgroup tile: 128 pixels x 64 channels; 8 wave32s in 4x2.
// Wave tile: 32x32 = 2x2 fragments; K unrolled by 64 (2 WMMA K-steps).
// LDS tiles are staged directly in WMMA fragment order so each fragment
// is one 32-byte contiguous LDS read per lane (ds_load_b128 pairs).
// ---------------------------------------------------------------------------
template <int KK>
__global__ __launch_bounds__(256) void conv_bn_wmma(
    const float* __restrict__ x, const float* __restrict__ w,
    const float* __restrict__ convb, const float* __restrict__ gamma,
    const float* __restrict__ beta, float* __restrict__ out,
    int NB, int Ci, int H, int W, int Co, int relu)
{
  const int K  = Ci * KK;
  const int HW = H * W;
  const int P  = NB * HW;

  const int tileM = blockIdx.x * 128;
  const int tileN = blockIdx.y * 64;

  // fragment-order LDS: [ksub][frag][lane][16 bf16]
  __shared__ __attribute__((aligned(32))) unsigned short As[2][8][32][16]; // 16 KB
  __shared__ __attribute__((aligned(32))) unsigned short Bs[2][4][32][16]; //  8 KB

  const int tid  = (int)threadIdx.x;
  const int lane = tid & 31;
  const int wave = tid >> 5;
  const int wm   = wave >> 1;   // 0..3 : pixel subtile (32 pixels each)
  const int wn   = wave & 1;    // 0..1 : channel subtile (32 channels each)

  // Per-thread im2col pixel decomposition (loop-invariant; hoisted out of K loop)
  const int arow  = tid >> 1;   // 0..127 : pixel row in tile
  const int ahalf = tid & 1;    // ki parity
  const int p = tileM + arow;
  const bool pvalid = p < P;
  int n = 0, oy = 0, ox = 0;
  if (pvalid) {
    n = p / HW;
    int rm = p - n * HW;
    oy = rm / W;
    ox = rm - oy * W;
  }
  const float* xn = x + (long)n * Ci * HW;

  v8f acc[2][2];
#pragma unroll
  for (int mf = 0; mf < 2; ++mf)
#pragma unroll
    for (int nf = 0; nf < 2; ++nf)
#pragma unroll
      for (int r = 0; r < 8; ++r) acc[mf][nf][r] = 0.0f;

  const int mfg = arow >> 4, ml = arow & 15;

  for (int k0 = 0; k0 < K; k0 += 64) {
    // ---- stage A (im2col), 32 elements / thread, fragment-order ----
#pragma unroll 4
    for (int j = 0; j < 32; ++j) {
      int ki = ahalf + (j << 1);
      int k  = k0 + ki;
      float v = 0.0f;
      if (pvalid && k < K) {
        int ci, kh, kw;
        if (KK == 9) {
          ci = k / 9;
          int kr = k - ci * 9;
          kh = kr / 3;
          kw = kr - kh * 3;
        } else {
          ci = k; kh = 0; kw = 0;
        }
        int iy = oy + kh - (KK == 9 ? 1 : 0);
        int ix = ox + kw - (KK == 9 ? 1 : 0);
        if (iy >= 0 && iy < H && ix >= 0 && ix < W)
          v = xn[(long)ci * HW + iy * W + ix];
      }
      int ks = ki >> 5, kl = ki & 31;
      As[ks][mfg][ml + ((kl & 8) ? 16 : 0)][(kl & 7) + ((kl & 16) ? 8 : 0)] = f2bf(v);
    }
    // ---- stage B (weights) 64x64, 16 / thread, coalesced along k ----
#pragma unroll 4
    for (int i2 = tid; i2 < 64 * 64; i2 += 256) {
      int ni = i2 >> 6, ki = i2 & 63;
      int k = k0 + ki, co = tileN + ni;
      float v = (k < K && co < Co) ? w[(long)co * K + k] : 0.0f;
      int ks = ki >> 5, kl = ki & 31;
      Bs[ks][ni >> 4][(ni & 15) + ((kl & 8) ? 16 : 0)][(kl & 7) + ((kl & 16) ? 8 : 0)] = f2bf(v);
    }
    if (k0 + 64 < K && tid < 64 && (tileN + tid) < Co)
      __builtin_prefetch(&w[(long)(tileN + tid) * K + k0 + 64], 0, 1);
    __syncthreads();

    // ---- 8 WMMAs per barrier interval ----
#pragma unroll
    for (int ks = 0; ks < 2; ++ks) {
      v16bf a0 = *(const v16bf*)As[ks][(wm << 1) + 0][lane];
      v16bf a1 = *(const v16bf*)As[ks][(wm << 1) + 1][lane];
      v16bf b0 = *(const v16bf*)Bs[ks][(wn << 1) + 0][lane];
      v16bf b1 = *(const v16bf*)Bs[ks][(wn << 1) + 1][lane];
      acc[0][0] = WMMA_BF16(a0, b0, acc[0][0]);
      acc[0][1] = WMMA_BF16(a0, b1, acc[0][1]);
      acc[1][0] = WMMA_BF16(a1, b0, acc[1][0]);
      acc[1][1] = WMMA_BF16(a1, b1, acc[1][1]);
    }
    __syncthreads();
  }

  // ---- epilogue: BN fold + optional ReLU, write NCHW fp32 ----
#pragma unroll
  for (int nf = 0; nf < 2; ++nf) {
    int col = tileN + (wn << 5) + (nf << 4) + (lane & 15);
    if (col < Co) {
      float sc = gamma[col] * BN_SCALE;
      float bs = convb[col] * sc + beta[col];
#pragma unroll
      for (int mf = 0; mf < 2; ++mf) {
#pragma unroll
        for (int r = 0; r < 8; ++r) {
          int rowp = tileM + (wm << 5) + (mf << 4) + r + ((lane < 16) ? 0 : 8);
          if (rowp < P) {
            float v = acc[mf][nf][r] * sc + bs;
            if (relu) v = fmaxf(v, 0.0f);
            int nn = rowp / HW;
            int rm = rowp - nn * HW;
            out[((long)nn * Co + col) * HW + rm] = v;
          }
        }
      }
    }
  }
}

// ---------------------------------------------------------------------------
// out = maxpool2x2( relu(conv2 + idn) )
// ---------------------------------------------------------------------------
__global__ void res_relu_pool(const float* __restrict__ conv2,
                              const float* __restrict__ idn,
                              float* __restrict__ out,
                              int NB, int C, int H, int W)
{
  int Ho = H >> 1, Wo = W >> 1;
  long total = (long)NB * C * Ho * Wo;
  for (long i = (long)blockIdx.x * blockDim.x + threadIdx.x; i < total;
       i += (long)gridDim.x * blockDim.x) {
    long t = i;
    int wo = (int)(t % Wo); t /= Wo;
    int ho = (int)(t % Ho); t /= Ho;
    int c  = (int)(t % C);
    int nn = (int)(t / C);
    long base = (((long)nn * C + c) * H + (ho << 1)) * W + (wo << 1);
    float v00 = fmaxf(conv2[base]         + idn[base],         0.0f);
    float v01 = fmaxf(conv2[base + 1]     + idn[base + 1],     0.0f);
    float v10 = fmaxf(conv2[base + W]     + idn[base + W],     0.0f);
    float v11 = fmaxf(conv2[base + W + 1] + idn[base + W + 1], 0.0f);
    out[i] = fmaxf(fmaxf(v00, v01), fmaxf(v10, v11));
  }
}

// ---------------------------------------------------------------------------
// Generic WMMA GEMM: out[row*ldo+col] = act(A(MxK) @ W(KxN) + bias + addC)
// act: 0 none, 1 tanh.  Workgroup tile 16 x 256, wave tile 16 x 32 (2 frags),
// K unrolled by 64 -> 4 WMMAs per barrier interval.
// ---------------------------------------------------------------------------
__global__ __launch_bounds__(256) void gemm_bias_act(
    const float* __restrict__ A, const float* __restrict__ Wt,
    const float* __restrict__ bias, const float* __restrict__ addC,
    float* __restrict__ out, int M, int K, int N, int ldo, int act)
{
  __shared__ __attribute__((aligned(32))) unsigned short As[2][1][32][16];  //  2 KB
  __shared__ __attribute__((aligned(32))) unsigned short Bs[2][16][32][16]; // 32 KB

  const int tid  = (int)threadIdx.x;
  const int lane = tid & 31;
  const int wave = tid >> 5;
  const int m0 = blockIdx.x * 16;
  const int n0 = blockIdx.y * 256;

  v8f acc0, acc1;
#pragma unroll
  for (int r = 0; r < 8; ++r) { acc0[r] = 0.0f; acc1[r] = 0.0f; }

  for (int k0 = 0; k0 < K; k0 += 64) {
    // ---- stage A 16x64, 4 / thread ----
#pragma unroll 4
    for (int i = tid; i < 16 * 64; i += 256) {
      int mi = i >> 6, ki = i & 63;
      float v = (m0 + mi < M && k0 + ki < K) ? A[(long)(m0 + mi) * K + k0 + ki] : 0.0f;
      int ks = ki >> 5, kl = ki & 31;
      As[ks][0][(mi & 15) + ((kl & 8) ? 16 : 0)][(kl & 7) + ((kl & 16) ? 8 : 0)] = f2bf(v);
    }
    // ---- stage B 64x256, 64 / thread, coalesced along n ----
#pragma unroll 8
    for (int i = tid; i < 64 * 256; i += 256) {
      int ki = i >> 8, ni = i & 255;
      float v = (k0 + ki < K && n0 + ni < N) ? Wt[(long)(k0 + ki) * N + n0 + ni] : 0.0f;
      int ks = ki >> 5, kl = ki & 31;
      Bs[ks][ni >> 4][(ni & 15) + ((kl & 8) ? 16 : 0)][(kl & 7) + ((kl & 16) ? 8 : 0)] = f2bf(v);
    }
    if (k0 + 64 < K && (n0 + tid) < N)
      __builtin_prefetch(&Wt[(long)(k0 + 64) * N + n0 + tid], 0, 1);
    __syncthreads();

#pragma unroll
    for (int ks = 0; ks < 2; ++ks) {
      v16bf a  = *(const v16bf*)As[ks][0][lane];
      v16bf b0 = *(const v16bf*)Bs[ks][(wave << 1) + 0][lane];
      v16bf b1 = *(const v16bf*)Bs[ks][(wave << 1) + 1][lane];
      acc0 = WMMA_BF16(a, b0, acc0);
      acc1 = WMMA_BF16(a, b1, acc1);
    }
    __syncthreads();
  }

#pragma unroll
  for (int nf = 0; nf < 2; ++nf) {
    int col = n0 + (wave << 5) + (nf << 4) + (lane & 15);
    if (col < N) {
      float bv = bias ? bias[col] : 0.0f;
#pragma unroll
      for (int r = 0; r < 8; ++r) {
        int row = m0 + r + ((lane < 16) ? 0 : 8);
        if (row < M) {
          float v = (nf == 0 ? acc0[r] : acc1[r]) + bv;
          if (addC) v += addC[(long)row * N + col];
          if (act == 1) v = tanhf(v);
          out[(long)row * ldo + col] = v;
        }
      }
    }
  }
}

// ---------------------------------------------------------------------------
// f (16,512,7,7) -> featsT (16*49, 512) : featsT[b*49+l][c] = f[b][c][l]
// ---------------------------------------------------------------------------
__global__ void transpose_feats(const float* __restrict__ f, float* __restrict__ o)
{
  int i = blockIdx.x * blockDim.x + threadIdx.x;
  if (i < 16 * 49 * 512) {
    int c = i & 511;
    int t = i >> 9;
    int l = t % 49;
    int b = t / 49;
    o[i] = f[((long)b * 512 + c) * 49 + l];
  }
}

// feats (16,49,256) -> m (16,256)
__global__ void mean_feats(const float* __restrict__ feats, float* __restrict__ m)
{
  int i = blockIdx.x * blockDim.x + threadIdx.x;
  if (i < 16 * 256) {
    int b = i >> 8, c = i & 255;
    float s = 0.0f;
    for (int l = 0; l < 49; ++l) s += feats[((long)b * 49 + l) * 256 + c];
    m[i] = s * (1.0f / 49.0f);
  }
}

// emb[b][t][e] = table[captions[b][t]][e], t in [0,20)
__global__ void embed_gather(const int* __restrict__ caps,
                             const float* __restrict__ table,
                             float* __restrict__ emb)
{
  int i = blockIdx.x * blockDim.x + threadIdx.x;
  if (i < 16 * 20 * 256) {
    int e = i & 255;
    int t = (i >> 8) % 20;
    int b = i / (20 * 256);
    int tok = caps[b * 21 + t];
    emb[i] = table[(long)tok * 256 + e];
  }
}

// ---------------------------------------------------------------------------
// Per-batch attention step (one workgroup per batch element)
// ---------------------------------------------------------------------------
__global__ __launch_bounds__(256) void attention_step(
    const float* __restrict__ enc_proj, const float* __restrict__ dec,
    const float* __restrict__ full_w, const float* __restrict__ full_b,
    const float* __restrict__ feats, const float* __restrict__ emb_all, int t,
    float* __restrict__ xbuf, float* __restrict__ alphas_out)
{
  __shared__ float e_sh[49];
  __shared__ float a_sh[49];

  const int b = blockIdx.x;
  const int tid = (int)threadIdx.x;
  const int lane = tid & 31;
  const int wave = tid >> 5;

  for (int l = wave; l < 49; l += 8) {
    float s = 0.0f;
    for (int k = lane; k < 256; k += 32) {
      float v = tanhf(enc_proj[((long)b * 49 + l) * 256 + k] + dec[b * 256 + k]);
      s += v * full_w[k];
    }
#pragma unroll
    for (int o = 16; o; o >>= 1) s += __shfl_xor(s, o, 32);
    if (lane == 0) e_sh[l] = s + full_b[0];
  }
  __syncthreads();

  if (wave == 0) {
    float mx = -1e30f;
    for (int l = lane; l < 49; l += 32) mx = fmaxf(mx, e_sh[l]);
#pragma unroll
    for (int o = 16; o; o >>= 1) mx = fmaxf(mx, __shfl_xor(mx, o, 32));
    float sum = 0.0f;
    for (int l = lane; l < 49; l += 32) {
      float ex = __expf(e_sh[l] - mx);
      a_sh[l] = ex;
      sum += ex;
    }
#pragma unroll
    for (int o = 16; o; o >>= 1) sum += __shfl_xor(sum, o, 32);
    float inv = 1.0f / sum;
    for (int l = lane; l < 49; l += 32) {
      float al = a_sh[l] * inv;
      a_sh[l] = al;
      alphas_out[((long)b * 49 + l) * 20 + t] = al;
    }
  }
  __syncthreads();

  float ctx = 0.0f;
  for (int l = 0; l < 49; ++l) ctx += a_sh[l] * feats[((long)b * 49 + l) * 256 + tid];
  xbuf[b * 512 + 256 + tid] = ctx;
  xbuf[b * 512 + tid]       = emb_all[((long)b * 20 + t) * 256 + tid];
}

// LSTM pointwise: gates (16,2048) order [i|f|g|o]; updates h,c (16,512)
__global__ void lstm_cell(const float* __restrict__ gates,
                          float* __restrict__ h, float* __restrict__ c)
{
  int i = blockIdx.x * blockDim.x + threadIdx.x;
  if (i < 16 * 512) {
    int b = i >> 9, k = i & 511;
    const float* gr = gates + (long)b * 2048;
    float ig = sigmoidf_(gr[k]);
    float fg = sigmoidf_(gr[512 + k]);
    float gg = tanhf(gr[1024 + k]);
    float og = sigmoidf_(gr[1536 + k]);
    float cn = fg * c[i] + ig * gg;
    c[i] = cn;
    h[i] = og * tanhf(cn);
  }
}

// ---------------------------------------------------------------------------
// Host orchestration
// ---------------------------------------------------------------------------
extern "C" void kernel_launch(void* const* d_in, const int* in_sizes, int n_in,
                              void* d_out, int out_size, void* d_ws, size_t ws_size,
                              hipStream_t stream)
{
  (void)in_sizes; (void)n_in; (void)out_size; (void)ws_size;

  const float* images   = (const float*)d_in[0];
  const int*   captions = (const int*)d_in[1];

  const float* embed_w    = (const float*)d_in[58];
  const float* embed_b    = (const float*)d_in[59];
  const float* emb_table  = (const float*)d_in[60];
  const float* att_enc_w  = (const float*)d_in[61];
  const float* att_enc_b  = (const float*)d_in[62];
  const float* att_dec_w  = (const float*)d_in[63];
  const float* att_dec_b  = (const float*)d_in[64];
  const float* att_full_w = (const float*)d_in[65];
  const float* att_full_b = (const float*)d_in[66];
  const float* w_ih       = (const float*)d_in[67];
  const float* b_ih       = (const float*)d_in[68];
  const float* w_hh       = (const float*)d_in[69];
  const float* b_hh       = (const float*)d_in[70];
  const float* init_h_w   = (const float*)d_in[71];
  const float* init_h_b   = (const float*)d_in[72];
  const float* init_c_w   = (const float*)d_in[73];
  const float* init_c_b   = (const float*)d_in[74];
  const float* fc_w       = (const float*)d_in[75];
  const float* fc_b       = (const float*)d_in[76];

  float* logits_out = (float*)d_out;                  // (16,20,5000)
  float* alphas_out = (float*)d_out + 16 * 20 * 5000; // (16,49,20)

  // ---- workspace carve ----
  size_t off = 0;
  char* base = (char*)d_ws;
  auto wsalloc = [&](size_t nfloat) -> float* {
    float* p = (float*)(base + off);
    off += ((nfloat * sizeof(float) + 255) / 256) * 256;
    return p;
  };
  float* conv1buf = wsalloc(51380224);  // 16*64*224*224 (max pre-pool)
  float* conv2buf = wsalloc(51380224);
  float* skipbuf  = wsalloc(51380224);
  float* pool0    = wsalloc(12845056);  // 16*64*112*112 (max pooled)
  float* pool1    = wsalloc(12845056);
  float* featsT   = wsalloc(784 * 512);
  float* feats    = wsalloc(784 * 256);
  float* encp     = wsalloc(784 * 256);
  float* mbuf     = wsalloc(16 * 256);
  float* hbuf     = wsalloc(16 * 512);
  float* cbuf     = wsalloc(16 * 512);
  float* decb     = wsalloc(16 * 256);
  float* xbuf     = wsalloc(16 * 512);
  float* tmpg     = wsalloc(16 * 2048);
  float* gates    = wsalloc(16 * 2048);
  float* embA     = wsalloc(16 * 20 * 256);

#ifdef HAVE_TDM
  tdm_null_probe<<<1, 32, 0, stream>>>();
#endif

  // ---- CNN encoder ----
  const int chans[5][4] = {
      {3, 64, 224, 224}, {64, 128, 112, 112}, {128, 256, 56, 56},
      {256, 512, 28, 28}, {512, 512, 14, 14}};
  const float* curx = images;
  float* pools[2] = {pool0, pool1};
  int pidx = 0;
  int pbase = 2;
  for (int bi = 0; bi < 5; ++bi) {
    int Ci = chans[bi][0], Co = chans[bi][1], H = chans[bi][2], W = chans[bi][3];
    bool hasSkip = (Ci != Co);
    const float* c1w = (const float*)d_in[pbase + 0];
    const float* c1b = (const float*)d_in[pbase + 1];
    const float* g1  = (const float*)d_in[pbase + 2];
    const float* bb1 = (const float*)d_in[pbase + 3];
    const float* c2w = (const float*)d_in[pbase + 4];
    const float* c2b = (const float*)d_in[pbase + 5];
    const float* g2  = (const float*)d_in[pbase + 6];
    const float* bb2 = (const float*)d_in[pbase + 7];

    int P = 16 * H * W;
    dim3 grd((P + 127) / 128, (Co + 63) / 64);
    conv_bn_wmma<9><<<grd, 256, 0, stream>>>(curx, c1w, c1b, g1, bb1, conv1buf,
                                             16, Ci, H, W, Co, 1);
    conv_bn_wmma<9><<<grd, 256, 0, stream>>>(conv1buf, c2w, c2b, g2, bb2, conv2buf,
                                             16, Co, H, W, Co, 0);
    const float* idn = curx;
    if (hasSkip) {
      const float* skw    = (const float*)d_in[pbase + 8];
      const float* skb    = (const float*)d_in[pbase + 9];
      const float* skg    = (const float*)d_in[pbase + 10];
      const float* skbeta = (const float*)d_in[pbase + 11];
      conv_bn_wmma<1><<<grd, 256, 0, stream>>>(curx, skw, skb, skg, skbeta, skipbuf,
                                               16, Ci, H, W, Co, 0);
      idn = skipbuf;
    }
    long total = (long)16 * Co * (H / 2) * (W / 2);
    int pgrid = (int)((total + 255) / 256);
    res_relu_pool<<<pgrid, 256, 0, stream>>>(conv2buf, idn, pools[pidx], 16, Co, H, W);
    curx = pools[pidx];
    pidx ^= 1;
    pbase += hasSkip ? 12 : 8;
  }

  // ---- decoder setup ----
  transpose_feats<<<(16 * 49 * 512 + 255) / 256, 256, 0, stream>>>(curx, featsT);
  gemm_bias_act<<<dim3(49, 1), 256, 0, stream>>>(featsT, embed_w, embed_b, nullptr,
                                                 feats, 784, 512, 256, 256, 0);
  gemm_bias_act<<<dim3(49, 1), 256, 0, stream>>>(feats, att_enc_w, att_enc_b, nullptr,
                                                 encp, 784, 256, 256, 256, 0);
  mean_feats<<<(16 * 256 + 255) / 256, 256, 0, stream>>>(feats, mbuf);
  gemm_bias_act<<<dim3(1, 2), 256, 0, stream>>>(mbuf, init_h_w, init_h_b, nullptr,
                                                hbuf, 16, 256, 512, 512, 1);
  gemm_bias_act<<<dim3(1, 2), 256, 0, stream>>>(mbuf, init_c_w, init_c_b, nullptr,
                                                cbuf, 16, 256, 512, 512, 1);
  embed_gather<<<(16 * 20 * 256 + 255) / 256, 256, 0, stream>>>(captions, emb_table, embA);

  // ---- 20 LSTM-attention steps ----
  for (int t = 0; t < 20; ++t) {
    gemm_bias_act<<<dim3(1, 1), 256, 0, stream>>>(hbuf, att_dec_w, att_dec_b, nullptr,
                                                  decb, 16, 512, 256, 256, 0);
    attention_step<<<16, 256, 0, stream>>>(encp, decb, att_full_w, att_full_b,
                                           feats, embA, t, xbuf, alphas_out);
    gemm_bias_act<<<dim3(1, 8), 256, 0, stream>>>(xbuf, w_ih, b_ih, nullptr,
                                                  tmpg, 16, 512, 2048, 2048, 0);
    gemm_bias_act<<<dim3(1, 8), 256, 0, stream>>>(hbuf, w_hh, b_hh, tmpg,
                                                  gates, 16, 512, 2048, 2048, 0);
    lstm_cell<<<(16 * 512 + 255) / 256, 256, 0, stream>>>(gates, hbuf, cbuf);
    gemm_bias_act<<<dim3(1, 20), 256, 0, stream>>>(hbuf, fc_w, fc_b, nullptr,
                                                   logits_out + (long)t * 5000,
                                                   16, 512, 5000, 20 * 5000, 0);
  }
}
